// AttackGraphGNN_85272280695061
// MI455X (gfx1250) — matrix operand order repaired and checked by reading, not compiled, not run on hardware
//
#include <hip/hip_runtime.h>
#include <hip/hip_bf16.h>

#define NN     20000
#define FIN    64
#define HID    128
#define NHEAD  4
#define EE     320000
#define ETOT   (EE + NN)        // edges + self loops
#define NEG_SLOPE 0.2f

typedef __attribute__((ext_vector_type(16))) __bf16 v16bf;
typedef __attribute__((ext_vector_type(8)))  __bf16 v8bf;
typedef __attribute__((ext_vector_type(8)))  float  v8f;

// ---------------------------------------------------------------------------
// order-preserving float <-> uint mapping for atomicMax-based segment max
// ---------------------------------------------------------------------------
__device__ __forceinline__ unsigned fkey(float f) {
  unsigned u = __float_as_uint(f);
  return (u & 0x80000000u) ? ~u : (u | 0x80000000u);
}
__device__ __forceinline__ float funkey(unsigned k) {
  unsigned u = (k & 0x80000000u) ? (k & 0x7FFFFFFFu) : ~k;
  return __uint_as_float(u);
}

// ---------------------------------------------------------------------------
// elementwise f32 -> bf16 conversion (A-side operands, row-major)
// ---------------------------------------------------------------------------
__global__ void cvt_bf16_kernel(const float* __restrict__ in,
                                __bf16* __restrict__ out, int n) {
  int i = blockIdx.x * blockDim.x + threadIdx.x;
  if (i < n) out[i] = (__bf16)in[i];
}

// ---------------------------------------------------------------------------
// pack a row-major f32 B[K x NC] into bf16 WMMA-fragment order:
//   out[(((kb*TILESN + tn)*32 + lane)*16 + i] = B[(kb*32 + bko + i)*NC + col]
// with col = tn*16 + (lane&15), bko = (lane<16 ? 0 : 16).
// Each lane's 16-value fragment then sits in 32 contiguous bytes.
// ---------------------------------------------------------------------------
template <int NC>
__global__ void pack_b_bf16(const float* __restrict__ B,
                            __bf16* __restrict__ out, int K) {
  const int idx = blockIdx.x * blockDim.x + threadIdx.x;
  if (idx >= K * NC) return;
  constexpr int TILESN = NC / 16;
  const int i    = idx & 15;
  const int lane = (idx >> 4) & 31;
  const int rest = idx >> 9;                 // kb*TILESN + tn
  const int tn   = rest % TILESN;
  const int kb   = rest / TILESN;
  const int col  = tn * 16 + (lane & 15);
  const int k    = kb * 32 + ((lane < 16) ? 0 : 16) + i;
  out[idx] = (__bf16)B[k * NC + col];
}

__global__ void fill_u32_kernel(unsigned* __restrict__ p, unsigned v, int n) {
  int i = blockIdx.x * blockDim.x + threadIdx.x;
  if (i < n) p[i] = v;
}

// ---------------------------------------------------------------------------
// bf16 WMMA GEMM: C[M x NC] = act(A[M x K] @ B + bias)
// A row-major bf16, B pre-packed in fragment order. One wave per 16x16 tile,
// fully unrolled K -> K/32 v_wmma_f32_16x16x32_bf16 per tile.
// ---------------------------------------------------------------------------
template <int K, int NC, bool BIAS, bool RELU, bool MIRROR>
__global__ void gemm_wmma(const __bf16* __restrict__ A,
                          const __bf16* __restrict__ Bp,
                          const float* __restrict__ bias,
                          float* __restrict__ C,
                          __bf16* __restrict__ Cb, int ntiles) {
  constexpr int TILESN  = NC / 16;
  constexpr int KBSTRID = TILESN * 32 * 16;      // elements per k-block of Bp
  const int wave = threadIdx.x >> 5;
  const int lane = threadIdx.x & 31;
  const int tile = blockIdx.x * (blockDim.x >> 5) + wave;
  if (tile >= ntiles) return;                    // wave-uniform exit

  const int tm = tile / TILESN;
  const int tn = tile % TILESN;
  const int row0 = tm << 4;

  // A fragment (16x32 bf16): lanes 0-15 row=lane K 0-15;
  // lanes 16-31 same rows, K 16-31.  Two 16B contiguous runs.
  const int arow = row0 + (lane & 15);
  const int alo  = (lane < 16) ? 0 : 16;
  const __bf16* Ap = A + (size_t)arow * K;
  const __bf16* Bl = Bp + ((size_t)tn * 32 + lane) * 16;

  v8f acc = {};
#pragma unroll
  for (int kb = 0; kb < K / 32; ++kb) {
    const v8bf a0 = *(const v8bf*)(Ap + kb * 32 + alo);
    const v8bf a1 = *(const v8bf*)(Ap + kb * 32 + alo + 8);
    const v16bf a = __builtin_shufflevector(a0, a1, 0, 1, 2, 3, 4, 5, 6, 7, 8,
                                            9, 10, 11, 12, 13, 14, 15);
    const v16bf b = *(const v16bf*)(Bl + (size_t)kb * KBSTRID);
    acc = __builtin_amdgcn_wmma_f32_16x16x32_bf16(
        false, a, false, b, (short)0, acc, false, false);
  }

  // C layout: VGPR r -> M = row0 + r (lanes 0-15) / row0 + r + 8 (lanes 16-31)
  const int cm0 = row0 + ((lane < 16) ? 0 : 8);
  const int cn  = (tn << 4) + (lane & 15);
  const float bv = BIAS ? bias[cn] : 0.0f;
#pragma unroll
  for (int r = 0; r < 8; ++r) {
    float v = acc[r] + bv;
    if (RELU) v = fmaxf(v, 0.0f);
    const int idx = (cm0 + r) * NC + cn;
    C[idx] = v;
    if (MIRROR) Cb[idx] = (__bf16)v;
  }
}

// ---------------------------------------------------------------------------
// per-(node,head) attention coefficients: a = <h[n,hd,:], att[hd,:]>
// ---------------------------------------------------------------------------
__global__ void attn_coeff_kernel(const float* __restrict__ h,
                                  const float* __restrict__ att_src,
                                  const float* __restrict__ att_dst,
                                  float* __restrict__ a_src,
                                  float* __restrict__ a_dst) {
  const int wid  = (blockIdx.x * blockDim.x + threadIdx.x) >> 5;
  const int lane = threadIdx.x & 31;
  if (wid >= NN * NHEAD) return;
  const int n  = wid >> 2;
  const int hd = wid & 3;
  const float* hp = h + (size_t)n * (NHEAD * HID) + hd * HID;
  const float* as = att_src + hd * HID;
  const float* ad = att_dst + hd * HID;
  float s1 = 0.0f, s2 = 0.0f;
#pragma unroll
  for (int d0 = 0; d0 < HID; d0 += 32) {
    const float hv = hp[d0 + lane];
    s1 += hv * as[d0 + lane];
    s2 += hv * ad[d0 + lane];
  }
  for (int off = 16; off > 0; off >>= 1) {
    s1 += __shfl_down(s1, off);
    s2 += __shfl_down(s2, off);
  }
  if (lane == 0) { a_src[wid] = s1; a_dst[wid] = s2; }
}

__device__ __forceinline__ void edge_endpoints(const long long* ei, int e,
                                               int& s, int& d) {
  if (e < EE) { s = (int)ei[e]; d = (int)ei[EE + e]; }
  else        { s = e - EE;     d = e - EE; }            // self loop
}

__device__ __forceinline__ float leaky(float v) {
  return v > 0.0f ? v : v * NEG_SLOPE;
}

// pass 1: segment max over destinations (ordered-uint atomicMax)
__global__ void edge_max_kernel(const long long* __restrict__ ei,
                                const float* __restrict__ a_src,
                                const float* __restrict__ a_dst,
                                unsigned* __restrict__ mkey) {
  const int t = blockIdx.x * blockDim.x + threadIdx.x;
  if (t >= ETOT * NHEAD) return;
  const int e = t >> 2, hd = t & 3;
  int s, d; edge_endpoints(ei, e, s, d);
  const float ev = leaky(a_src[s * NHEAD + hd] + a_dst[d * NHEAD + hd]);
  atomicMax(&mkey[d * NHEAD + hd], fkey(ev));
}

// pass 2: w = exp(e - m[dst]); z[dst] += w
__global__ void edge_expsum_kernel(const long long* __restrict__ ei,
                                   const float* __restrict__ a_src,
                                   const float* __restrict__ a_dst,
                                   const unsigned* __restrict__ mkey,
                                   float* __restrict__ wbuf,
                                   float* __restrict__ z) {
  const int t = blockIdx.x * blockDim.x + threadIdx.x;
  if (t >= ETOT * NHEAD) return;
  const int e = t >> 2, hd = t & 3;
  int s, d; edge_endpoints(ei, e, s, d);
  const float ev = leaky(a_src[s * NHEAD + hd] + a_dst[d * NHEAD + hd]);
  const float m  = funkey(mkey[d * NHEAD + hd]);
  const float w  = __expf(ev - m);
  wbuf[t] = w;
  atomicAdd(&z[d * NHEAD + hd], w);
}

// pass 3: head-folded scatter. msg[d] = 0.25 * sum_h alpha_h * h[src,h,d]
// (mean over heads is linear -> fold before scatter: 4x fewer atomics)
__global__ void edge_scatter_kernel(const long long* __restrict__ ei,
                                    const float* __restrict__ h,
                                    const float* __restrict__ wbuf,
                                    const float* __restrict__ z,
                                    float* __restrict__ acc) {
  const int e    = (blockIdx.x * blockDim.x + threadIdx.x) >> 5;
  const int lane = threadIdx.x & 31;
  if (e >= ETOT) return;
  int s, d; edge_endpoints(ei, e, s, d);
  const float a0 = 0.25f * wbuf[e * NHEAD + 0] / z[d * NHEAD + 0];
  const float a1 = 0.25f * wbuf[e * NHEAD + 1] / z[d * NHEAD + 1];
  const float a2 = 0.25f * wbuf[e * NHEAD + 2] / z[d * NHEAD + 2];
  const float a3 = 0.25f * wbuf[e * NHEAD + 3] / z[d * NHEAD + 3];
  const float* hp = h + (size_t)s * (NHEAD * HID);
  float* ap = acc + (size_t)d * HID;
#pragma unroll
  for (int k0 = 0; k0 < HID; k0 += 32) {
    const int k = k0 + lane;
    const float msg = a0 * hp[k] + a1 * hp[HID + k] +
                      a2 * hp[2 * HID + k] + a3 * hp[3 * HID + k];
    atomicAdd(&ap[k], msg);
  }
}

// residual update: x += relu(acc + bias); refresh bf16 mirror
__global__ void node_update_kernel(float* __restrict__ x,
                                   __bf16* __restrict__ xb,
                                   const float* __restrict__ acc,
                                   const float* __restrict__ bias) {
  const int i = blockIdx.x * blockDim.x + threadIdx.x;
  if (i >= NN * HID) return;
  const int d = i & (HID - 1);
  const float v = x[i] + fmaxf(acc[i] + bias[d], 0.0f);
  x[i] = v;
  xb[i] = (__bf16)v;
}

// ---------------------------------------------------------------------------
// prediction head: one wave per node
// ---------------------------------------------------------------------------
__global__ void pred_head_kernel(const float* __restrict__ x,
                                 const float* __restrict__ W1,
                                 const float* __restrict__ b1,
                                 const float* __restrict__ W2,
                                 const float* __restrict__ b2,
                                 const float* __restrict__ vW,
                                 const float* __restrict__ vb,
                                 float* __restrict__ out) {
  const int n    = (blockIdx.x * blockDim.x + threadIdx.x) >> 5;
  const int lane = threadIdx.x & 31;
  if (n >= NN) return;
  const float* xp = x + (size_t)n * HID;
  float h1a = 0.0f, h1b = 0.0f, vs = 0.0f;
  for (int d = 0; d < HID; ++d) {
    const float xv = xp[d];
    h1a += xv * W1[d * 64 + lane];
    h1b += xv * W1[d * 64 + lane + 32];
  }
  h1a = fmaxf(h1a + b1[lane], 0.0f);
  h1b = fmaxf(h1b + b1[lane + 32], 0.0f);
  float p = h1a * W2[lane] + h1b * W2[lane + 32];
#pragma unroll
  for (int d0 = 0; d0 < HID; d0 += 32) vs += xp[d0 + lane] * vW[d0 + lane];
  for (int off = 16; off > 0; off >>= 1) {
    p  += __shfl_down(p, off);
    vs += __shfl_down(vs, off);
  }
  if (lane == 0) {
    out[n]      = 1.0f / (1.0f + __expf(-(p + b2[0])));
    out[NN + n] = 1.0f / (1.0f + __expf(-(vs + vb[0])));
  }
}

// ---------------------------------------------------------------------------
extern "C" void kernel_launch(void* const* d_in, const int* in_sizes, int n_in,
                              void* d_out, int out_size, void* d_ws, size_t ws_size,
                              hipStream_t stream) {
  const float*     node_features = (const float*)d_in[0];
  const long long* edge_index    = (const long long*)d_in[1];
  const float*     enc_W   = (const float*)d_in[2];
  const float*     enc_b   = (const float*)d_in[3];
  const float*     gat_lin = (const float*)d_in[4];
  const float*     att_src = (const float*)d_in[5];
  const float*     att_dst = (const float*)d_in[6];
  const float*     gat_bias = (const float*)d_in[7];
  const float*     pred_W1 = (const float*)d_in[8];
  const float*     pred_b1 = (const float*)d_in[9];
  const float*     pred_W2 = (const float*)d_in[10];
  const float*     pred_b2 = (const float*)d_in[11];
  const float*     vuln_W  = (const float*)d_in[12];
  const float*     vuln_b  = (const float*)d_in[13];
  float* out = (float*)d_out;

  // ---- workspace bump allocator (all L2-resident: ~76 MB total) ----
  char* ws = (char*)d_ws;
  auto alloc = [&](size_t bytes) -> char* {
    char* p = ws;
    ws += (bytes + 255) & ~(size_t)255;
    return p;
  };
  float*    x     = (float*)alloc((size_t)NN * HID * 4);
  __bf16*   xb    = (__bf16*)alloc((size_t)NN * HID * 2);
  float*    h     = (float*)alloc((size_t)NN * NHEAD * HID * 4);
  float*    a_src = (float*)alloc((size_t)NN * NHEAD * 4);
  float*    a_dst = (float*)alloc((size_t)NN * NHEAD * 4);
  unsigned* mkey  = (unsigned*)alloc((size_t)NN * NHEAD * 4);
  float*    z     = (float*)alloc((size_t)NN * NHEAD * 4);
  float*    wbuf  = (float*)alloc((size_t)ETOT * NHEAD * 4);
  float*    accb  = (float*)alloc((size_t)NN * HID * 4);
  __bf16*   nf_b   = (__bf16*)alloc((size_t)NN * FIN * 2);
  __bf16*   encW_p = (__bf16*)alloc((size_t)FIN * HID * 2);           // packed
  __bf16*   linW_p = (__bf16*)alloc((size_t)3 * HID * NHEAD * HID * 2); // packed

  const int TPB = 256;
  auto nb = [](long long n, int t) { return (int)((n + t - 1) / t); };

  // ---- bf16 staging: A-side row-major, B-side packed into fragment order --
  cvt_bf16_kernel<<<nb((long long)NN * FIN, TPB), TPB, 0, stream>>>(
      node_features, nf_b, NN * FIN);
  pack_b_bf16<HID><<<nb((long long)FIN * HID, TPB), TPB, 0, stream>>>(
      enc_W, encW_p, FIN);
  for (int l = 0; l < 3; ++l) {
    pack_b_bf16<NHEAD * HID>
        <<<nb((long long)HID * NHEAD * HID, TPB), TPB, 0, stream>>>(
            gat_lin + (size_t)l * HID * NHEAD * HID,
            linW_p + (size_t)l * HID * NHEAD * HID, HID);
  }

  // ---- encoder: x = relu(nf @ enc_W + enc_b), with bf16 mirror ----
  {
    const int tiles = (NN / 16) * (HID / 16);       // 10000
    gemm_wmma<FIN, HID, true, true, true><<<tiles / 8, TPB, 0, stream>>>(
        nf_b, encW_p, enc_b, x, xb, tiles);
  }

  // ---- 3 GAT layers ----
  for (int l = 0; l < 3; ++l) {
    constexpr int NC = NHEAD * HID;                 // 512
    const int tiles = (NN / 16) * (NC / 16);        // 40000
    gemm_wmma<HID, NC, false, false, false><<<tiles / 8, TPB, 0, stream>>>(
        xb, linW_p + (size_t)l * HID * NC, nullptr, h, nullptr, tiles);

    attn_coeff_kernel<<<nb((long long)NN * NHEAD * 32, TPB), TPB, 0, stream>>>(
        h, att_src + (size_t)l * NHEAD * HID, att_dst + (size_t)l * NHEAD * HID,
        a_src, a_dst);

    fill_u32_kernel<<<nb(NN * NHEAD, TPB), TPB, 0, stream>>>(mkey, 0u, NN * NHEAD);
    fill_u32_kernel<<<nb(NN * NHEAD, TPB), TPB, 0, stream>>>((unsigned*)z, 0u, NN * NHEAD);
    fill_u32_kernel<<<nb((long long)NN * HID, TPB), TPB, 0, stream>>>(
        (unsigned*)accb, 0u, NN * HID);

    edge_max_kernel<<<nb((long long)ETOT * NHEAD, TPB), TPB, 0, stream>>>(
        edge_index, a_src, a_dst, mkey);
    edge_expsum_kernel<<<nb((long long)ETOT * NHEAD, TPB), TPB, 0, stream>>>(
        edge_index, a_src, a_dst, mkey, wbuf, z);
    edge_scatter_kernel<<<nb((long long)ETOT * 32, TPB), TPB, 0, stream>>>(
        edge_index, h, wbuf, z, accb);

    node_update_kernel<<<nb((long long)NN * HID, TPB), TPB, 0, stream>>>(
        x, xb, accb, gat_bias + (size_t)l * HID);
  }

  // ---- prediction head ----
  pred_head_kernel<<<nb((long long)NN * 32, TPB), TPB, 0, stream>>>(
      x, pred_W1, pred_b1, pred_W2, pred_b2, vuln_W, vuln_b, out);
}
